// BaseCrossAttention_62878321214089
// MI455X (gfx1250) — compile-verified, matrix operand order checked by reading, not run on hardware
//
#include <hip/hip_runtime.h>
#include <hip/hip_bf16.h>

typedef _Float16 f16;
typedef __attribute__((ext_vector_type(16))) _Float16 v16h;
typedef __attribute__((ext_vector_type(8)))  _Float16 v8h;
typedef __attribute__((ext_vector_type(8)))  float    v8f;

// ---------------------------------------------------------------------------
// CDNA5 async global->LDS copy (ASYNCcnt-tracked). Per lane: 16 bytes from the
// lane's global address to the lane's LDS offset. LDS offset = low 32 bits of
// the generic shared-memory pointer (ISA 10.2: LDS aperture uses addr[31:0]).
// ---------------------------------------------------------------------------
static __device__ __forceinline__ void async_b128(const void* g, const void* l) {
    asm volatile("global_load_async_to_lds_b128 %0, %1, off"
                 :
                 : "v"((unsigned)(size_t)l), "v"(g)
                 : "memory");
}
static __device__ __forceinline__ void wait_async_le(int) {}
static __device__ __forceinline__ void wait_async_4() {
    asm volatile("s_wait_asynccnt 0x4" ::: "memory");
}
static __device__ __forceinline__ void wait_async_0() {
    asm volatile("s_wait_asynccnt 0x0" ::: "memory");
}

// Build a WMMA 16-bit operand fragment (v16h) from two contiguous 8xf16 chunks.
// Per CDNA5 ISA layout: elements 0-7 = K base+0..7, elements 8-15 = K base+16..23
// (the +8*half K offset is applied by the caller in the pointer).
static __device__ __forceinline__ v16h make_frag(const f16* lo, const f16* hi) {
    v8h a = *(const v8h*)lo;
    v8h b = *(const v8h*)hi;
    v16h r;
#pragma unroll
    for (int i = 0; i < 8; ++i) { r[i] = a[i]; r[i + 8] = b[i]; }
    return r;
}

// ---------------------------------------------------------------------------
// Weight transpose + f32->f16 convert:  Wt[n*K + k] = (f16)W[k*N + n]
// ---------------------------------------------------------------------------
__global__ __launch_bounds__(256)
void wtrans_kernel(const float* __restrict__ W, f16* __restrict__ Wt, int K, int N) {
    size_t idx = (size_t)blockIdx.x * 256 + threadIdx.x;
    if (idx >= (size_t)K * N) return;
    int n = (int)(idx / K);
    int k = (int)(idx % K);
    Wt[idx] = (f16)W[(size_t)k * N + n];
}

// ---------------------------------------------------------------------------
// Row LayerNorm: f32 in -> f16 out, one 256-thread block per row.
// ---------------------------------------------------------------------------
__global__ __launch_bounds__(256)
void ln_rows_kernel(const float* __restrict__ X, const float* __restrict__ g,
                    const float* __restrict__ bb, f16* __restrict__ Yo, int D) {
    __shared__ float r1[256];
    __shared__ float r2[256];
    const int row = blockIdx.x, tid = threadIdx.x;
    const float* xr = X + (size_t)row * D;
    float s = 0.f, q = 0.f;
    for (int i = tid; i < D; i += 256) { float v = xr[i]; s += v; q += v * v; }
    r1[tid] = s; r2[tid] = q;
    __syncthreads();
    for (int o = 128; o > 0; o >>= 1) {
        if (tid < o) { r1[tid] += r1[tid + o]; r2[tid] += r2[tid + o]; }
        __syncthreads();
    }
    float mean = r1[0] / D;
    float var  = r2[0] / D - mean * mean;
    float rs   = rsqrtf(var + 1e-5f);
    for (int i = tid; i < D; i += 256)
        Yo[(size_t)row * D + i] = (f16)((xr[i] - mean) * rs * g[i] + bb[i]);
}

// ---------------------------------------------------------------------------
// Stylization: h = silu( LN(y)*(1+scale[b]) + shift[b] ), f16 in -> f16 out.
// ss = (B, 2D): scale = ss[b, 0:D], shift = ss[b, D:2D]
// ---------------------------------------------------------------------------
__global__ __launch_bounds__(256)
void stylize_kernel(const f16* __restrict__ Yin, const float* __restrict__ g,
                    const float* __restrict__ bb, const float* __restrict__ ss,
                    f16* __restrict__ Ho, int D, int T) {
    __shared__ float r1[256];
    __shared__ float r2[256];
    const int row = blockIdx.x, tid = threadIdx.x;
    const int b = row / T;
    const f16* yr = Yin + (size_t)row * D;
    float s = 0.f, q = 0.f;
    for (int i = tid; i < D; i += 256) { float v = (float)yr[i]; s += v; q += v * v; }
    r1[tid] = s; r2[tid] = q;
    __syncthreads();
    for (int o = 128; o > 0; o >>= 1) {
        if (tid < o) { r1[tid] += r1[tid + o]; r2[tid] += r2[tid + o]; }
        __syncthreads();
    }
    float mean = r1[0] / D;
    float var  = r2[0] / D - mean * mean;
    float rs   = rsqrtf(var + 1e-5f);
    const float* sc = ss + (size_t)b * 2 * D;
    for (int i = tid; i < D; i += 256) {
        float v = ((float)yr[i] - mean) * rs * g[i] + bb[i];
        v = v * (1.0f + sc[i]) + sc[D + i];
        v = v / (1.0f + __expf(-v));   // silu
        Ho[(size_t)row * D + i] = (f16)v;
    }
}

// ---------------------------------------------------------------------------
// Small GEMM: ss[b, n] = sum_k silu(emb[b,k]) * We[k,n] + be[n]   (32 x 2048 x 2048)
// ---------------------------------------------------------------------------
__global__ __launch_bounds__(256)
void emb_gemm_kernel(const float* __restrict__ emb, const float* __restrict__ We,
                     const float* __restrict__ be, float* __restrict__ ss) {
    const int n = blockIdx.x * 256 + threadIdx.x;  // 0..2047
    const int b = blockIdx.y;
    const float* er = emb + (size_t)b * 2048;
    float acc = 0.f;
    for (int k = 0; k < 2048; ++k) {
        float e = er[k];
        float sv = e / (1.0f + __expf(-e));
        acc += sv * We[(size_t)k * 2048 + n];
    }
    ss[(size_t)b * 2048 + n] = acc + be[n];
}

// ---------------------------------------------------------------------------
// WMMA GEMM with double-buffered async global->LDS staging.
// C(MxN) = A(MxK,f16 row-major) @ Bt(NxK,f16 row-major)^T
// Block tile 128x128, 8 waves (4 in M x 2 in N), wave tile 32x64 = 2x4 WMMA.
// Epilogue: +bias[n], optional *tcond(row batch), optional +residual -> f32 out,
// else f16 out.  Requires M%128==0, N%128==0, K%32==0 (true for all uses).
// ---------------------------------------------------------------------------
#define BM 128
#define BN 128
#define BK 32

__global__ __launch_bounds__(256)
void gemm_f16_kernel(const f16* __restrict__ A, const f16* __restrict__ Bt,
                     const float* __restrict__ bias,
                     f16* __restrict__ Cf16, float* __restrict__ Cf32,
                     const float* __restrict__ residual,
                     const int* __restrict__ cond, int rowsPerBatch,
                     int M, int N, int K) {
    __shared__ __align__(16) f16 As[2][BM][BK + 8];
    __shared__ __align__(16) f16 Bs[2][BN][BK + 8];
    const int tid  = threadIdx.x;
    const int wave = tid >> 5, lane = tid & 31;
    const int half = lane >> 4, l15 = lane & 15;
    const int wm = wave >> 1, wn = wave & 1;
    const size_t blockM = (size_t)blockIdx.y * BM;
    const size_t blockN = (size_t)blockIdx.x * BN;

    const int r0 = tid >> 2;         // 0..63
    const int cg = (tid & 3) * 8;    // 0,8,16,24

    // Issue one 128x32 A tile + 128x32 B tile as async copies (4 insts/thread).
    auto issue_tile = [&](int k0, int buf) {
#pragma unroll
        for (int it = 0; it < 2; ++it) {
            int r = r0 + it * 64;
            async_b128(&A[(blockM + r) * K + k0 + cg], &As[buf][r][cg]);
            async_b128(&Bt[(blockN + r) * K + k0 + cg], &Bs[buf][r][cg]);
        }
    };

    v8f acc[2][4];
    v8f zero = {};
#pragma unroll
    for (int mi = 0; mi < 2; ++mi)
#pragma unroll
        for (int ni = 0; ni < 4; ++ni) acc[mi][ni] = zero;

    const int nIter = K / BK;
    issue_tile(0, 0);
    for (int i = 0; i < nIter; ++i) {
        const int cur = i & 1;
        if (i + 1 < nIter) {
            issue_tile((i + 1) * BK, cur ^ 1);  // prefetch next tile
            wait_async_4();                      // first 4 (current tile) done
        } else {
            wait_async_0();
        }
        __syncthreads();                         // all waves' tile in LDS

        v16h af[2], bf[4];
#pragma unroll
        for (int mi = 0; mi < 2; ++mi) {
            int row = wm * 32 + mi * 16 + l15;
            af[mi] = make_frag(&As[cur][row][half * 8], &As[cur][row][16 + half * 8]);
        }
#pragma unroll
        for (int ni = 0; ni < 4; ++ni) {
            int col = wn * 64 + ni * 16 + l15;
            bf[ni] = make_frag(&Bs[cur][col][half * 8], &Bs[cur][col][16 + half * 8]);
        }
#pragma unroll
        for (int mi = 0; mi < 2; ++mi)
#pragma unroll
            for (int ni = 0; ni < 4; ++ni)
                acc[mi][ni] = __builtin_amdgcn_wmma_f32_16x16x32_f16(
                    false, af[mi], false, bf[ni], (short)0, acc[mi][ni], false, false);
        __syncthreads();                         // reads done -> buffer reusable
    }

    // Epilogue. C/D layout: element j -> row M = j + 8*half, col N = l15.
#pragma unroll
    for (int mi = 0; mi < 2; ++mi) {
#pragma unroll
        for (int ni = 0; ni < 4; ++ni) {
            size_t col = blockN + wn * 64 + ni * 16 + l15;
            float bcol = bias ? bias[col] : 0.0f;
#pragma unroll
            for (int j = 0; j < 8; ++j) {
                size_t row = blockM + wm * 32 + mi * 16 + j + 8 * half;
                float v = acc[mi][ni][j] + bcol;
                if (cond) {
                    int bidx = (int)(row / (size_t)rowsPerBatch);
                    v *= (cond[bidx] % 10 > 0) ? 1.0f : 0.0f;
                }
                if (Cf32) {
                    size_t idx = row * (size_t)N + col;
                    Cf32[idx] = v + (residual ? residual[idx] : 0.0f);
                } else {
                    Cf16[row * (size_t)N + col] = (f16)v;
                }
            }
        }
    }
}

// ---------------------------------------------------------------------------
// Attention per (b,h): scores = Q Kt (no 1/sqrt(d) scale per reference),
// softmax over N=256, y = P V.  Block: 256 threads = 8 waves, each wave owns
// 16 t-rows; block covers 128 t-rows.  grid = (B*H, T/128).
// Dynamic LDS: kt[256][72] f16 (K head tile, [n][d]); ps[8*16][264] f16 probs;
// vs[64][40] f16 (V chunk transposed, aliases kt after phase 1).  ~102 KB.
// ---------------------------------------------------------------------------
__global__ __launch_bounds__(256)
void attn_kernel(const f16* __restrict__ Q, const f16* __restrict__ Kf,
                 const f16* __restrict__ Vf, f16* __restrict__ Y) {
    extern __shared__ __align__(16) char smem_raw[];
    f16* kt = (f16*)smem_raw;            // 256*72
    f16* ps = kt + 256 * 72;             // 128*264
    f16* vs = kt;                        // alias: used after kt is dead

    const int tid  = threadIdx.x;
    const int wave = tid >> 5, lane = tid & 31;
    const int half = lane >> 4, l15 = lane & 15;
    const int bh = blockIdx.x;
    const int b  = bh >> 4;   // H = 16
    const int h  = bh & 15;
    const int t0 = blockIdx.y * 128;

    // ---- async-load K head tile into LDS: kt[n][d], n=0..255, d=0..63 ----
    {
        int n = tid;
        const f16* src = Kf + ((size_t)b * 256 + n) * 1024 + h * 64;
#pragma unroll
        for (int dg = 0; dg < 8; ++dg)
            async_b128(&src[dg * 8], &kt[n * 72 + dg * 8]);
        wait_async_0();
    }
    __syncthreads();

    // ---- phase 1: scores 16 x 256 per wave (16 n-tiles, K=64 in 2 steps) ----
    const f16* qrow =
        Q + ((size_t)b * 1024 + (t0 + wave * 16 + l15)) * 1024 + h * 64;
    v16h aq[2];
#pragma unroll
    for (int ks = 0; ks < 2; ++ks) {
        int d0 = ks * 32;
        aq[ks] = make_frag(&qrow[d0 + half * 8], &qrow[d0 + 16 + half * 8]);
    }
    v8f sc[16];
    v8f zero = {};
#pragma unroll
    for (int nt = 0; nt < 16; ++nt) sc[nt] = zero;
#pragma unroll
    for (int nt = 0; nt < 16; ++nt) {
        int n = nt * 16 + l15;
#pragma unroll
        for (int ks = 0; ks < 2; ++ks) {
            int d0 = ks * 32;
            v16h bk = make_frag(&kt[n * 72 + d0 + half * 8],
                                &kt[n * 72 + d0 + 16 + half * 8]);
            sc[nt] = __builtin_amdgcn_wmma_f32_16x16x32_f16(
                false, aq[ks], false, bk, (short)0, sc[nt], false, false);
        }
    }

    // ---- softmax over N within wave; rows live as (j + 8*half), col = l15 ----
#pragma unroll
    for (int j = 0; j < 8; ++j) {
        float m = -1e30f;
#pragma unroll
        for (int nt = 0; nt < 16; ++nt) m = fmaxf(m, sc[nt][j]);
#pragma unroll
        for (int o = 1; o < 16; o <<= 1) m = fmaxf(m, __shfl_xor(m, o, 32));
        float ssum = 0.f;
#pragma unroll
        for (int nt = 0; nt < 16; ++nt) {
            float e = __expf(sc[nt][j] - m);
            sc[nt][j] = e;
            ssum += e;
        }
#pragma unroll
        for (int o = 1; o < 16; o <<= 1) ssum += __shfl_xor(ssum, o, 32);
        float inv = 1.0f / ssum;
        int r = j + 8 * half;
#pragma unroll
        for (int nt = 0; nt < 16; ++nt)
            ps[(wave * 16 + r) * 264 + nt * 16 + l15] = (f16)(sc[nt][j] * inv);
    }
    __syncthreads();   // ps complete; kt dead -> vs may reuse its space

    // ---- phase 2: y = P V, K=256 in 8 chunks of 32 ----
    v8f ya[4];
#pragma unroll
    for (int dt = 0; dt < 4; ++dt) ya[dt] = zero;

    for (int n0 = 0; n0 < 256; n0 += 32) {
        {   // load V chunk transposed: vs[d][nn], d=0..63, nn=0..31
            int nn = tid >> 3;
            int dg = (tid & 7) * 8;
            const f16* src =
                Vf + ((size_t)b * 256 + n0 + nn) * 1024 + h * 64 + dg;
            v8h vv = *(const v8h*)src;
#pragma unroll
            for (int q2 = 0; q2 < 8; ++q2) vs[(dg + q2) * 40 + nn] = vv[q2];
        }
        __syncthreads();
        v16h ap = make_frag(&ps[(wave * 16 + l15) * 264 + n0 + half * 8],
                            &ps[(wave * 16 + l15) * 264 + n0 + 16 + half * 8]);
#pragma unroll
        for (int dt = 0; dt < 4; ++dt) {
            int d = dt * 16 + l15;
            v16h bv = make_frag(&vs[d * 40 + half * 8],
                                &vs[d * 40 + 16 + half * 8]);
            ya[dt] = __builtin_amdgcn_wmma_f32_16x16x32_f16(
                false, ap, false, bv, (short)0, ya[dt], false, false);
        }
        __syncthreads();
    }

    // ---- write y: element j -> t-row j + 8*half, col = dt*16 + l15 ----
    f16* yout = Y + ((size_t)b * 1024 + t0 + wave * 16) * 1024 + h * 64;
#pragma unroll
    for (int dt = 0; dt < 4; ++dt)
#pragma unroll
        for (int j = 0; j < 8; ++j) {
            int r = j + 8 * half;
            yout[(size_t)r * 1024 + dt * 16 + l15] = (f16)ya[dt][j];
        }
}

// ---------------------------------------------------------------------------
// Host launch
// ---------------------------------------------------------------------------
extern "C" void kernel_launch(void* const* d_in, const int* in_sizes, int n_in,
                              void* d_out, int out_size, void* d_ws, size_t ws_size,
                              hipStream_t stream) {
    const int Bb = 32, T = 1024, D = 1024, Nn = 256, L = 768;
    const int MT = Bb * T;   // 32768
    const int MN = Bb * Nn;  // 8192

    const float* x      = (const float*)d_in[0];
    const float* xf     = (const float*)d_in[1];
    const float* emb    = (const float*)d_in[2];
    // d_in[3] = src_mask (all ones, unused by reference math)
    const int*   cond   = (const int*)d_in[4];
    const float* ln_x_g = (const float*)d_in[5];
    const float* ln_x_b = (const float*)d_in[6];
    const float* ln_t_g = (const float*)d_in[7];
    const float* ln_t_b = (const float*)d_in[8];
    const float* Wq     = (const float*)d_in[9];
    const float* bq     = (const float*)d_in[10];
    const float* Wk     = (const float*)d_in[11];
    const float* bk     = (const float*)d_in[12];
    const float* Wv     = (const float*)d_in[13];
    const float* bv     = (const float*)d_in[14];
    const float* We     = (const float*)d_in[15];
    const float* be     = (const float*)d_in[16];
    const float* ln_y_g = (const float*)d_in[17];
    const float* ln_y_b = (const float*)d_in[18];
    const float* Wo     = (const float*)d_in[19];
    const float* bo     = (const float*)d_in[20];
    float* out = (float*)d_out;

    char* ws = (char*)d_ws;
    size_t off = 0;
    auto alloc = [&](size_t bytes) -> char* {
        char* p = ws + off;
        off += (bytes + 255) & ~(size_t)255;
        return p;
    };
    f16* xln  = (f16*)alloc((size_t)MT * D * 2);   // LN(x); later reused as y
    f16* qbuf = (f16*)alloc((size_t)MT * D * 2);   // Q; later reused as h_in
    f16* xfln = (f16*)alloc((size_t)MN * L * 2);   // LN(xf)
    f16* wqT  = (f16*)alloc((size_t)D * D * 2);
    f16* wkT  = (f16*)alloc((size_t)D * L * 2);
    f16* wvT  = (f16*)alloc((size_t)D * L * 2);
    f16* woT  = (f16*)alloc((size_t)D * D * 2);
    f16* kbuf = (f16*)alloc((size_t)MN * D * 2);
    f16* vbuf = (f16*)alloc((size_t)MN * D * 2);
    float* ssb = (float*)alloc((size_t)Bb * 2 * D * 4);
    f16* ybuf = xln;    // reuse
    f16* hbuf = qbuf;   // reuse

    // 1) weight transposes (f32 -> f16, [N][K])
    wtrans_kernel<<<(D * D + 255) / 256, 256, 0, stream>>>(Wq, wqT, D, D);
    wtrans_kernel<<<(L * D + 255) / 256, 256, 0, stream>>>(Wk, wkT, L, D);
    wtrans_kernel<<<(L * D + 255) / 256, 256, 0, stream>>>(Wv, wvT, L, D);
    wtrans_kernel<<<(D * D + 255) / 256, 256, 0, stream>>>(Wo, woT, D, D);

    // 2) LayerNorms
    ln_rows_kernel<<<MT, 256, 0, stream>>>(x, ln_x_g, ln_x_b, xln, D);
    ln_rows_kernel<<<MN, 256, 0, stream>>>(xf, ln_t_g, ln_t_b, xfln, L);

    // 3) projections (WMMA GEMMs with async double-buffered staging)
    gemm_f16_kernel<<<dim3(D / BN, MT / BM), 256, 0, stream>>>(
        xln, wqT, bq, qbuf, nullptr, nullptr, nullptr, 0, MT, D, D);
    gemm_f16_kernel<<<dim3(D / BN, MN / BM), 256, 0, stream>>>(
        xfln, wkT, bk, kbuf, nullptr, nullptr, nullptr, 0, MN, D, L);
    gemm_f16_kernel<<<dim3(D / BN, MN / BM), 256, 0, stream>>>(
        xfln, wvT, bv, vbuf, nullptr, nullptr, cond, Nn, MN, D, L);

    // 4) emb path: scale/shift
    emb_gemm_kernel<<<dim3(2 * D / 256, Bb), 256, 0, stream>>>(emb, We, be, ssb);

    // 5) attention (dynamic LDS: kt + ps)
    const int attn_lds = (256 * 72 + 128 * 264) * (int)sizeof(f16);  // 104448 B
    attn_kernel<<<dim3(Bb * 16, T / 128), 256, attn_lds, stream>>>(
        qbuf, kbuf, vbuf, ybuf);

    // 6) stylization -> h_in (f16)
    stylize_kernel<<<MT, 256, 0, stream>>>(ybuf, ln_y_g, ln_y_b, ssb, hbuf, D, T);

    // 7) output projection + residual -> d_out (f32)
    gemm_f16_kernel<<<dim3(D / BN, MT / BM), 256, 0, stream>>>(
        hbuf, woT, bo, nullptr, out, x, nullptr, 0, MT, D, D);
}
